// CrossSelfAttention_77266461655383
// MI455X (gfx1250) — matrix operand order repaired and checked by reading, not compiled, not run on hardware
//
#include <hip/hip_runtime.h>
#include <hip/hip_bf16.h>

typedef __bf16 bf16_t;
typedef __bf16 v16bf __attribute__((ext_vector_type(16)));
typedef __bf16 v8bf  __attribute__((ext_vector_type(8)));
typedef float  v8f   __attribute__((ext_vector_type(8)));

#define NHEAD 12
#define CEMBD 768
#define TSEQ  1024
#define BSZ   8
#define HD    64
#define MTOK  (BSZ * TSEQ)   // 8192

// Load a 16-element bf16 fragment as two contiguous 8-element (16B) chunks.
__device__ __forceinline__ v16bf ld2x8(const bf16_t* p0, const bf16_t* p1) {
  union { v16bf v; v8bf h[2]; } u;
  u.h[0] = *(const v8bf*)p0;
  u.h[1] = *(const v8bf*)p1;
  return u.v;
}

__device__ __forceinline__ v8f wmma_bf16(v16bf a, v16bf b, v8f c) {
  // D = A(16x32 bf16) * B(32x16 bf16) + C(16x16 f32)
  return __builtin_amdgcn_wmma_f32_16x16x32_bf16(false, a, false, b, (short)0, c,
                                                 false, false);
}

// CDNA5 async memory->LDS copy (16B per lane), tracked with ASYNCcnt.
// the vdst VGPR gives the LDS address; low 32 bits of a flat shared pointer are the
// LDS offset, so truncating the generic pointer is the right operand.
__device__ __forceinline__ void async_copy16B(bf16_t* lptr, const bf16_t* gptr) {
  unsigned l = (unsigned)(uintptr_t)lptr;
  asm volatile("global_load_async_to_lds_b128 %0, %1, off"
               :
               : "v"(l), "v"(gptr)
               : "memory");
}
__device__ __forceinline__ void wait_async0() {
  asm volatile("s_wait_asynccnt 0" ::: "memory");
}

// ---------------- conversion kernels ----------------

__global__ void cvt_bf16_kernel(const float* __restrict__ in,
                                bf16_t* __restrict__ out, int n) {
  int i = (blockIdx.x * blockDim.x + threadIdx.x) * 4;
  if (i + 3 < n) {
    float4 f = *(const float4*)(in + i);
    out[i + 0] = (bf16_t)f.x;
    out[i + 1] = (bf16_t)f.y;
    out[i + 2] = (bf16_t)f.z;
    out[i + 3] = (bf16_t)f.w;
  }
}

// in: f32 [K][N] row-major  ->  out: bf16 [N][K] (transposed, K contiguous)
__global__ void transpose_bf16_kernel(const float* __restrict__ in,
                                      bf16_t* __restrict__ out, int K, int N) {
  int idx = blockIdx.x * blockDim.x + threadIdx.x;
  if (idx < K * N) {
    int k = idx / N;
    int n = idx - k * N;
    out[(size_t)n * K + k] = (bf16_t)in[idx];
  }
}

// ---------------- GEMM: C[M,N] = A[M,768] * W[768,N] + bias ----------------
// A: bf16 [M,768] row-major.  WT: bf16 [N,768] (W transposed, K contiguous).
// MODE 0: N=2304, scatter into q [B,H,T,64], k [B,H,T,64], vT [B,H,64,T] (bf16)
// MODE 1: N=768,  write f32 out [M,768]
template <int MODE>
__global__ __launch_bounds__(128) void gemm_kernel(
    const bf16_t* __restrict__ A, const bf16_t* __restrict__ WT,
    const float* __restrict__ bias, bf16_t* __restrict__ qb,
    bf16_t* __restrict__ kb, bf16_t* __restrict__ vtb,
    float* __restrict__ outf) {
  __shared__ bf16_t sA[64][40];  // 64 rows x 32 k (+8 pad)
  __shared__ bf16_t sW[64][40];  // 64 cols x 32 k (+8 pad)
  const int K = CEMBD;
  const int m0 = blockIdx.x * 64;
  const int n0 = blockIdx.y * 64;
  const int tid = threadIdx.x;
  const int lane = tid & 31, w = tid >> 5;
  const int wm = (w & 1) * 32, wn = (w >> 1) * 32;
  const int r = lane & 15, hi = lane >> 4, lo = hi * 8;

  v8f acc[2][2] = {};

  const int srow = tid >> 1, shalf = tid & 1;  // staging assignment
  for (int k0 = 0; k0 < K; k0 += 32) {
    const bf16_t* ga = A + (size_t)(m0 + srow) * K + k0 + shalf * 16;
    const bf16_t* gw = WT + (size_t)(n0 + srow) * K + k0 + shalf * 16;
    // Async global->LDS (no VGPR round-trip), 4 x 16B per thread.
    async_copy16B(&sA[srow][shalf * 16], ga);
    async_copy16B(&sA[srow][shalf * 16 + 8], ga + 8);
    async_copy16B(&sW[srow][shalf * 16], gw);
    async_copy16B(&sW[srow][shalf * 16 + 8], gw + 8);
    if (k0 + 32 < K) {  // gfx1250 global_prefetch_b8 of next tiles
      __builtin_prefetch(ga + 32, 0, 1);
      __builtin_prefetch(gw + 32, 0, 1);
    }
    wait_async0();
    __syncthreads();
    v16bf af[2], bf[2];
#pragma unroll
    for (int mi = 0; mi < 2; ++mi)
      af[mi] = ld2x8(&sA[wm + mi * 16 + r][lo], &sA[wm + mi * 16 + r][16 + lo]);
#pragma unroll
    for (int ni = 0; ni < 2; ++ni)
      bf[ni] = ld2x8(&sW[wn + ni * 16 + r][lo],
                     &sW[wn + ni * 16 + r][16 + lo]);
#pragma unroll
    for (int mi = 0; mi < 2; ++mi)
#pragma unroll
      for (int ni = 0; ni < 2; ++ni)
        acc[mi][ni] = wmma_bf16(af[mi], bf[ni], acc[mi][ni]);
    __syncthreads();
  }

  // Epilogue. D layout: vgpr rr holds element (row rr+8*hi, col lane&15).
#pragma unroll
  for (int mi = 0; mi < 2; ++mi) {
#pragma unroll
    for (int ni = 0; ni < 2; ++ni) {
      const int colg = n0 + wn + ni * 16 + r;
      const float bv = bias[colg];
#pragma unroll
      for (int rr = 0; rr < 8; ++rr) {
        const int rowg = m0 + wm + mi * 16 + rr + 8 * hi;
        const float val = acc[mi][ni][rr] + bv;
        if (MODE == 0) {
          const int sec = colg / CEMBD;
          const int cw = colg - sec * CEMBD;
          const int h = cw >> 6, d = cw & 63;
          const int b = rowg >> 10, t = rowg & 1023;
          const size_t hb = (size_t)(b * NHEAD + h);
          if (sec == 0)
            qb[(hb * TSEQ + t) * HD + d] = (bf16_t)val;
          else if (sec == 1)
            kb[(hb * TSEQ + t) * HD + d] = (bf16_t)val;
          else
            vtb[(hb * HD + d) * TSEQ + t] = (bf16_t)val;
        } else {
          outf[(size_t)rowg * CEMBD + colg] = val;
        }
      }
    }
  }
}

// ---------------- Flash attention ----------------
// Q,K: bf16 [B*H][T][64]; Vt: bf16 [B*H][64][T]; Y: bf16 [B][T][C]
// Block = 64 q rows (4 waves x 16). K-blocks of 32 columns, online softmax.
__global__ __launch_bounds__(128) void flash_kernel(
    const bf16_t* __restrict__ Q, const bf16_t* __restrict__ Kc,
    const bf16_t* __restrict__ Vt, bf16_t* __restrict__ Y) {
  __shared__ bf16_t Pl[4][16][40];  // per-wave P tile (16 x 32, +pad)
  const int qblk = blockIdx.x;  // 0..15
  const int bh = blockIdx.y;    // 0..95
  const int b = bh / NHEAD, h = bh % NHEAD;
  const int tid = threadIdx.x, lane = tid & 31, w = tid >> 5;
  const int r = lane & 15, hi = lane >> 4, lo = hi * 8;
  const int qbase = qblk * 64;
  const int tq = qbase + w * 16 + r;  // A-fragment row

  const bf16_t* Qh = Q + (size_t)bh * TSEQ * HD;
  const bf16_t* Kh = Kc + (size_t)bh * TSEQ * HD;
  const bf16_t* Vh = Vt + (size_t)bh * HD * TSEQ;

  // Q fragments resident for the whole kernel (two 32-wide d-chunks)
  v16bf qf[2];
#pragma unroll
  for (int c = 0; c < 2; ++c)
    qf[c] = ld2x8(&Qh[(size_t)tq * HD + c * 32 + lo],
                  &Qh[(size_t)tq * HD + c * 32 + 16 + lo]);

  float m_i[8], l_i[8];
  v8f o[4] = {};
#pragma unroll
  for (int i = 0; i < 8; ++i) {
    m_i[i] = -1e30f;
    l_i[i] = 0.f;
  }

  const float scale = 0.125f;  // 1/sqrt(64)
  const int nkb = qbase / 32 + 2;  // causal: cover k <= qbase+63 (uniform)

  for (int kb = 0; kb < nkb; ++kb) {
    const int k0 = kb * 32;
    // S = Q * K^T  (two 16-col tiles)
    v8f s[2] = {};
#pragma unroll
    for (int half = 0; half < 2; ++half) {
#pragma unroll
      for (int c = 0; c < 2; ++c) {
        const int krow = k0 + half * 16 + r;  // B-frag: N=lane&15 -> K row
        v16bf kf = ld2x8(&Kh[(size_t)krow * HD + c * 32 + lo],
                         &Kh[(size_t)krow * HD + c * 32 + 16 + lo]);
        s[half] = wmma_bf16(qf[c], kf, s[half]);
      }
    }
    // online softmax over the 32 new columns
#pragma unroll
    for (int rr = 0; rr < 8; ++rr) {
      const int tg = qbase + w * 16 + rr + 8 * hi;  // this row's token index
      float s0 = s[0][rr] * scale;
      float s1 = s[1][rr] * scale;
      if (k0 + r > tg) s0 = -1e30f;       // causal mask, col = k0 + N
      if (k0 + 16 + r > tg) s1 = -1e30f;  // col = k0 + 16 + N
      float mx = fmaxf(s0, s1);
#pragma unroll
      for (int off = 1; off < 16; off <<= 1)
        mx = fmaxf(mx, __shfl_xor(mx, off, 32));  // row max (16-lane half)
      const float mn = fmaxf(m_i[rr], mx);
      const float alpha = __expf(m_i[rr] - mn);
      const float p0 = (s0 > -1e29f) ? __expf(s0 - mn) : 0.f;
      const float p1 = (s1 > -1e29f) ? __expf(s1 - mn) : 0.f;
      float ps = p0 + p1;
#pragma unroll
      for (int off = 1; off < 16; off <<= 1) ps += __shfl_xor(ps, off, 32);
      l_i[rr] = l_i[rr] * alpha + ps;
      m_i[rr] = mn;
#pragma unroll
      for (int nt = 0; nt < 4; ++nt) o[nt][rr] *= alpha;
      Pl[w][rr + 8 * hi][r] = (bf16_t)p0;       // D layout -> LDS
      Pl[w][rr + 8 * hi][16 + r] = (bf16_t)p1;
    }
    __syncthreads();
    // P (A layout) and O += P * V
    v16bf pa = ld2x8(&Pl[w][r][lo], &Pl[w][r][16 + lo]);
#pragma unroll
    for (int nt = 0; nt < 4; ++nt) {
      v16bf vb = ld2x8(&Vh[(size_t)(nt * 16 + r) * TSEQ + k0 + lo],
                       &Vh[(size_t)(nt * 16 + r) * TSEQ + k0 + 16 + lo]);
      o[nt] = wmma_bf16(pa, vb, o[nt]);
    }
    __syncthreads();
  }

  // normalize and write y[b][t][h*64+d] as bf16 (input to proj GEMM)
#pragma unroll
  for (int rr = 0; rr < 8; ++rr) {
    const int tg = qbase + w * 16 + rr + 8 * hi;
    const float inv = 1.0f / l_i[rr];
#pragma unroll
    for (int nt = 0; nt < 4; ++nt) {
      const int d = nt * 16 + r;
      Y[((size_t)(b * TSEQ + tg)) * CEMBD + h * HD + d] =
          (bf16_t)(o[nt][rr] * inv);
    }
  }
}

// ---------------- launch ----------------

extern "C" void kernel_launch(void* const* d_in, const int* in_sizes, int n_in,
                              void* d_out, int out_size, void* d_ws,
                              size_t ws_size, hipStream_t stream) {
  const float* x      = (const float*)d_in[0];
  const float* w_attn = (const float*)d_in[1];
  const float* b_attn = (const float*)d_in[2];
  const float* w_proj = (const float*)d_in[3];
  const float* b_proj = (const float*)d_in[4];
  float* out = (float*)d_out;

  char* ws = (char*)d_ws;
  bf16_t* xb     = (bf16_t*)(ws + 0);          // 8192*768*2   = 12582912
  bf16_t* wattnT = (bf16_t*)(ws + 12582912);   // 2304*768*2   =  3538944
  bf16_t* wprojT = (bf16_t*)(ws + 16121856);   // 768*768*2    =  1179648
  bf16_t* qb     = (bf16_t*)(ws + 17301504);   // 96*1024*64*2 = 12582912
  bf16_t* kbuf   = (bf16_t*)(ws + 29884416);
  bf16_t* vtb    = (bf16_t*)(ws + 42467328);
  bf16_t* yb     = (bf16_t*)(ws + 55050240);   // total ~67.6 MB

  const int nx = MTOK * CEMBD;  // 6291456
  cvt_bf16_kernel<<<(nx / 4 + 255) / 256, 256, 0, stream>>>(x, xb, nx);
  const int nwa = CEMBD * 3 * CEMBD;
  transpose_bf16_kernel<<<(nwa + 255) / 256, 256, 0, stream>>>(w_attn, wattnT,
                                                               CEMBD, 3 * CEMBD);
  const int nwp = CEMBD * CEMBD;
  transpose_bf16_kernel<<<(nwp + 255) / 256, 256, 0, stream>>>(w_proj, wprojT,
                                                               CEMBD, CEMBD);
  // QKV projection: [8192,768] x [768,2304]
  gemm_kernel<0><<<dim3(MTOK / 64, (3 * CEMBD) / 64), 128, 0, stream>>>(
      xb, wattnT, b_attn, qb, kbuf, vtb, nullptr);
  // causal flash attention
  flash_kernel<<<dim3(TSEQ / 64, BSZ * NHEAD), 128, 0, stream>>>(qb, kbuf, vtb,
                                                                 yb);
  // output projection: [8192,768] x [768,768] -> f32 out
  gemm_kernel<1><<<dim3(MTOK / 64, CEMBD / 64), 128, 0, stream>>>(
      yb, wprojT, b_proj, nullptr, nullptr, nullptr, out);
}
